// Attention_67765993996325
// MI455X (gfx1250) — matrix operand order repaired and checked by reading, not compiled, not run on hardware
//
#include <hip/hip_runtime.h>

typedef __attribute__((ext_vector_type(16))) __bf16 v16bf;
typedef __attribute__((ext_vector_type(8)))  __bf16 v8bf;
typedef __attribute__((ext_vector_type(8)))  float  v8f;
typedef __attribute__((ext_vector_type(2)))  float  v2f;

#define WMMA_BF16(a, b, c) \
  __builtin_amdgcn_wmma_f32_16x16x32_bf16(false, (a), false, (b), (short)0, (c), false, false)
#define WMMA_F32(a, b, c) \
  __builtin_amdgcn_wmma_f32_16x16x4_f32(false, (a), false, (b), (short)0, (c), false, false)

// Problem constants
#define BATCH 8
#define CIN   64
#define NHEAD 8
#define DHEAD 64
#define NPIX  1024   // 32*32
#define WDIM  32
#define QSCALE 512.0f

// ---------------------------------------------------------------------------
// Kernel 1: QKV projection.  per-batch GEMM [1536 x 64] x [64 x 1024], f32 WMMA.
// One wave computes a 64x16 strip (4 o-tiles x 1 n-tile): the x (B) fragments
// are shared across the 4 o-tiles, so each strided x load feeds 4 WMMAs.
// Epilogue routes rows to Q (scaled, bf16 [n,d]), K^T (bf16 [d,n]), V (bf16 [n,d]);
// a 64-row o-group never crosses a 512-row section, so routing is wave-uniform.
// ---------------------------------------------------------------------------
__global__ __launch_bounds__(256) void qkv_proj(const float* __restrict__ x,
                                                const float* __restrict__ w,
                                                __bf16* __restrict__ qbf,
                                                __bf16* __restrict__ ktbf,
                                                __bf16* __restrict__ vbf) {
  const int wave = blockIdx.x * 8 + (threadIdx.x >> 5);
  const int lane = threadIdx.x & 31;
  const int nt   = wave & 63;        // 64 n-tiles
  const int tmp  = wave >> 6;
  const int og   = tmp % 24;         // 24 o-groups of 64 rows
  const int b    = tmp / 24;         // 8 batches
  const int o_base = og * 64, n0 = nt * 16;
  const int half = lane >> 4, lm = lane & 15;

  const float* xb = x + (size_t)b * CIN * NPIX;
  const float* wr0 = w + (o_base + lm) * CIN;     // A rows for the 4 o-tiles

  v8f acc[4] = {};
  #pragma unroll
  for (int k0 = 0; k0 < CIN; k0 += 4) {
    // B 4x16 f32 fragment (shared by all 4 o-tiles)
    v2f bfr;
    bfr.x = xb[(size_t)(k0 + half * 2 + 0) * NPIX + n0 + lm];
    bfr.y = xb[(size_t)(k0 + half * 2 + 1) * NPIX + n0 + lm];
    #pragma unroll
    for (int m = 0; m < 4; ++m) {
      v2f a = *(const v2f*)(wr0 + m * 16 * CIN + k0 + half * 2);
      acc[m] = WMMA_F32(a, bfr, acc[m]);
    }
  }

  const int sec = o_base >> 9;        // 0:q 1:k 2:v  (uniform across wave)
  const int nn  = n0 + lm;
  if (sec == 0) {
    #pragma unroll
    for (int m = 0; m < 4; ++m)
      #pragma unroll
      for (int r = 0; r < 8; ++r) {
        const int idx = (o_base & 511) + m * 16 + r + half * 8;
        const int h = idx >> 6, dk = idx & 63;
        qbf[((size_t)(b * NHEAD + h) * NPIX + nn) * DHEAD + dk] =
            (__bf16)(acc[m][r] * QSCALE);
      }
  } else if (sec == 1) {
    #pragma unroll
    for (int m = 0; m < 4; ++m)
      #pragma unroll
      for (int r = 0; r < 8; ++r) {
        const int idx = (o_base & 511) + m * 16 + r + half * 8;
        const int h = idx >> 6, dk = idx & 63;
        ktbf[((size_t)(b * NHEAD + h) * DHEAD + dk) * NPIX + nn] = (__bf16)acc[m][r];
      }
  } else {
    #pragma unroll
    for (int m = 0; m < 4; ++m)
      #pragma unroll
      for (int r = 0; r < 8; ++r) {
        const int idx = (o_base & 511) + m * 16 + r + half * 8;
        const int h = idx >> 6, dk = idx & 63;
        vbf[((size_t)(b * NHEAD + h) * NPIX + nn) * DHEAD + dk] = (__bf16)acc[m][r];
      }
  }
}

// ---------------------------------------------------------------------------
// Kernel 2: relative-position logits  logits[b,h,i,r] = q_i . rel[r,:]
// One wave per (bh, 16-row i-tile); r padded 63 -> 64.  bf16 WMMA.
// ---------------------------------------------------------------------------
__global__ __launch_bounds__(256) void rel_logits(const __bf16* __restrict__ qbf,
                                                  const float* __restrict__ rel_h,
                                                  const float* __restrict__ rel_w,
                                                  float* __restrict__ lh,
                                                  float* __restrict__ lw) {
  const int wave = blockIdx.x * 8 + (threadIdx.x >> 5);
  const int lane = threadIdx.x & 31;
  const int it = wave & 63;
  const int bh = wave >> 6;
  const int i0 = it * 16;
  const int half = lane >> 4, lm = lane & 15;

  // Q A-fragments (16x32 bf16), K-steps 0..31 / 32..63
  const __bf16* qrow = qbf + ((size_t)bh * NPIX + i0 + lm) * DHEAD;
  v16bf qa[2];
  #pragma unroll
  for (int s = 0; s < 2; ++s) {
    v8bf lo = *(const v8bf*)(qrow + s * 32 + half * 8);
    v8bf hi = *(const v8bf*)(qrow + s * 32 + 16 + half * 8);
    qa[s] = __builtin_shufflevector(lo, hi, 0, 1, 2, 3, 4, 5, 6, 7,
                                    8, 9, 10, 11, 12, 13, 14, 15);
  }

  const float* rel[2]  = {rel_h, rel_w};
  float*       outp[2] = {lh, lw};
  #pragma unroll
  for (int t = 0; t < 2; ++t) {
    const float* R = rel[t];
    float* dst = outp[t] + ((size_t)bh * NPIX + i0) * 64;
    #pragma unroll
    for (int rt = 0; rt < 4; ++rt) {            // r-tile: cols rt*16..+15
      v8f acc = {};
      #pragma unroll
      for (int s = 0; s < 2; ++s) {
        const int r = rt * 16 + lm;             // B: lane -> N column
        v16bf bf;
        #pragma unroll
        for (int e = 0; e < 16; ++e) {
          const int d = s * 32 + ((e < 8) ? half * 8 + e : 16 + half * 8 + (e - 8));
          bf[e] = (__bf16)((r < 63) ? R[r * DHEAD + d] : 0.0f);
        }
        acc = WMMA_BF16(qa[s], bf, acc);
      }
      #pragma unroll
      for (int rr = 0; rr < 8; ++rr)
        dst[(size_t)(rr + half * 8) * 64 + rt * 16 + lm] = acc[rr];
    }
  }
}

// ---------------------------------------------------------------------------
// Kernel 3: flash attention.  One wave = one (bh, 16-row q-tile).
// 32 j-iterations (one spatial row each -> rel_y constant per i-row, rel_x
// loop-invariant):
//   S = Q K^T (4 bf16 WMMAs), + bias, online softmax (shfl_xor in halves),
//   P (C-layout) -> LDS -> A-layout, O += P V (4 bf16 WMMAs).
// ---------------------------------------------------------------------------
__global__ __launch_bounds__(256) void attn_fwd(const __bf16* __restrict__ qbf,
                                                const __bf16* __restrict__ ktbf,
                                                const __bf16* __restrict__ vbf,
                                                const float* __restrict__ lh,
                                                const float* __restrict__ lw,
                                                float* __restrict__ out) {
  __shared__ __bf16 pLds[8][16][40];            // per-wave 16x32 staging (+pad)
  const int wslot = threadIdx.x >> 5;
  const int wave  = blockIdx.x * 8 + wslot;
  const int lane  = threadIdx.x & 31;
  const int qt = wave & 63;
  const int bh = wave >> 6;
  const int b = bh >> 3, h = bh & 7;
  const int i0 = qt * 16;
  const int half = lane >> 4, lm = lane & 15;

  // Q fragments
  const __bf16* qrow = qbf + ((size_t)bh * NPIX + i0 + lm) * DHEAD;
  v16bf qa[2];
  #pragma unroll
  for (int s = 0; s < 2; ++s) {
    v8bf lo = *(const v8bf*)(qrow + s * 32 + half * 8);
    v8bf hi = *(const v8bf*)(qrow + s * 32 + 16 + half * 8);
    qa[s] = __builtin_shufflevector(lo, hi, 0, 1, 2, 3, 4, 5, 6, 7,
                                    8, 9, 10, 11, 12, 13, 14, 15);
  }

  const __bf16* ktb = ktbf + (size_t)bh * DHEAD * NPIX;
  const __bf16* vb  = vbf + (size_t)bh * NPIX * DHEAD;
  const float* lhb = lh + (size_t)bh * NPIX * 64;
  const float* lwb = lw + (size_t)bh * NPIX * 64;

  int irow[8], yi[8];
  float mrow[8], lrow[8];
  float lw0[8], lw1[8];                 // rel_x bias: j-tile invariant
  #pragma unroll
  for (int r = 0; r < 8; ++r) {
    irow[r] = i0 + r + half * 8;
    yi[r] = irow[r] >> 5;
    const int xi = irow[r] & (WDIM - 1);
    lw0[r] = lwb[(size_t)irow[r] * 64 + (lm - xi + 31)];
    lw1[r] = lwb[(size_t)irow[r] * 64 + (16 + lm - xi + 31)];
    mrow[r] = -3.0e38f;
    lrow[r] = 0.0f;
  }
  v8f o_acc[4] = {};

  for (int j0 = 0; j0 < NPIX; j0 += 32) {
    // prefetch next tile's K/V rows (covers the serial softmax dependency)
    if (j0 + 32 < NPIX) {
      __builtin_prefetch(ktb + (size_t)lane * NPIX + j0 + 32, 0, 3);
      __builtin_prefetch(vb + (size_t)(j0 + 32 + lane) * DHEAD, 0, 3);
    }
    const int yj = j0 >> 5;                     // whole tile shares y_j
    // S = Q K^T : two 16x16 C tiles
    v8f s[2];
    #pragma unroll
    for (int t = 0; t < 2; ++t) {
      v8f acc = {};
      #pragma unroll
      for (int ks = 0; ks < 2; ++ks) {
        v16bf bf;
        #pragma unroll
        for (int e = 0; e < 16; ++e)
          bf[e] = ktb[(size_t)(ks * 32 + ((e < 8) ? half * 8 + e : 16 + half * 8 + (e - 8))) * NPIX + j0 + t * 16 + lm];
        acc = WMMA_BF16(qa[ks], bf, acc);
      }
      s[t] = acc;
    }
    // bias + online softmax
    #pragma unroll
    for (int r = 0; r < 8; ++r) {
      const float lhv = lhb[(size_t)irow[r] * 64 + (yj - yi[r] + 31)];
      s[0][r] += lhv + lw0[r];
      s[1][r] += lhv + lw1[r];

      float v = fmaxf(s[0][r], s[1][r]);
      #pragma unroll
      for (int m = 8; m >= 1; m >>= 1) v = fmaxf(v, __shfl_xor(v, m, 32));
      const float mnew  = fmaxf(mrow[r], v);
      const float alpha = __expf(mrow[r] - mnew);
      mrow[r] = mnew;
      lrow[r] *= alpha;
      #pragma unroll
      for (int t = 0; t < 4; ++t) o_acc[t][r] *= alpha;
      const float p0 = __expf(s[0][r] - mnew);
      const float p1 = __expf(s[1][r] - mnew);
      s[0][r] = p0; s[1][r] = p1;
      float rs = p0 + p1;
      #pragma unroll
      for (int m = 8; m >= 1; m >>= 1) rs += __shfl_xor(rs, m, 32);
      lrow[r] += rs;
    }
    // P: C-layout -> LDS -> A-layout (bf16).  Same-wave DS ops are in-order.
    #pragma unroll
    for (int t = 0; t < 2; ++t)
      #pragma unroll
      for (int r = 0; r < 8; ++r)
        pLds[wslot][r + half * 8][t * 16 + lm] = (__bf16)s[t][r];

    v16bf pa;
    {
      const __bf16* base = &pLds[wslot][lm][0];
      v8bf lo = *(const v8bf*)(base + half * 8);
      v8bf hi = *(const v8bf*)(base + 16 + half * 8);
      pa = __builtin_shufflevector(lo, hi, 0, 1, 2, 3, 4, 5, 6, 7,
                                   8, 9, 10, 11, 12, 13, 14, 15);
    }
    // O += P @ V
    #pragma unroll
    for (int t = 0; t < 4; ++t) {
      v16bf bv;
      #pragma unroll
      for (int e = 0; e < 16; ++e)
        bv[e] = vb[(size_t)(j0 + ((e < 8) ? half * 8 + e : 16 + half * 8 + (e - 8))) * DHEAD + t * 16 + lm];
      o_acc[t] = WMMA_BF16(pa, bv, o_acc[t]);
    }
  }

  // epilogue: normalize and store channel-major
  float* ob = out + ((size_t)b * 512 + h * DHEAD) * NPIX;
  float invl[8];
  #pragma unroll
  for (int r = 0; r < 8; ++r) invl[r] = 1.0f / lrow[r];
  #pragma unroll
  for (int t = 0; t < 4; ++t)
    #pragma unroll
    for (int r = 0; r < 8; ++r) {
      const int dk = t * 16 + lm;
      ob[(size_t)dk * NPIX + irow[r]] = o_acc[t][r] * invl[r];
    }
}

// ---------------------------------------------------------------------------
extern "C" void kernel_launch(void* const* d_in, const int* in_sizes, int n_in,
                              void* d_out, int out_size, void* d_ws, size_t ws_size,
                              hipStream_t stream) {
  const float* x    = (const float*)d_in[0];
  const float* wqkv = (const float*)d_in[1];
  const float* relh = (const float*)d_in[2];
  const float* relw = (const float*)d_in[3];
  float* out = (float*)d_out;

  char* ws = (char*)d_ws;
  __bf16* qbf  = (__bf16*)(ws + (size_t)0);          //  8 MiB  [64][1024][64]
  __bf16* ktbf = (__bf16*)(ws + ((size_t)8  << 20)); //  8 MiB  [64][64][1024]
  __bf16* vbf  = (__bf16*)(ws + ((size_t)16 << 20)); //  8 MiB  [64][1024][64]
  float*  lhp  = (float*) (ws + ((size_t)24 << 20)); // 16 MiB  [64][1024][64]
  float*  lwp  = (float*) (ws + ((size_t)40 << 20)); // 16 MiB

  // 8 batches * 24 o-groups * 64 n-tiles = 12288 waves / 8 per block
  qkv_proj<<<1536, 256, 0, stream>>>(x, wqkv, qbf, ktbf, vbf);
  // 64 bh * 64 i-tiles = 4096 waves
  rel_logits<<<512, 256, 0, stream>>>(qbf, relh, relw, lhp, lwp);
  // 64 bh * 64 q-tiles = 4096 waves
  attn_fwd<<<512, 256, 0, stream>>>(qbf, ktbf, vbf, lhp, lwp, out);
}
